// MoE_3478923510212
// MI455X (gfx1250) — compile-verified, matrix operand order checked by reading, not run on hardware
//
#include <hip/hip_runtime.h>
#include <hip/hip_bf16.h>

// Problem constants (B=2, S=2048, H=1024, F=2048, E=8, K=2)
#define T_TOK 4096
#define H_DIM 1024
#define E_NUM 8
#define F_DIM 2048
#define TWOF  4096
#define R_TOT 8192                 // T * K rows
#define BM    128                  // row tile
#define BN1   64                   // col tile, GEMM1 (per GLU half)
#define BN2   128                  // col tile, GEMM2
#define BK    32                   // K tile == WMMA K
#define LDP   48                   // LDS row pitch (elements; 96B: 16B-aligned, staggered banks)
#define MAX_TILES 72               // sum ceil(Ne/BM) <= 8192/128 + 8
#define R_PAD (MAX_TILES * BM)     // 9216 padded slots

#define AS1 __attribute__((address_space(1)))
#define AS3 __attribute__((address_space(3)))

typedef __attribute__((ext_vector_type(8)))  float          v8f;
typedef __attribute__((ext_vector_type(16))) __bf16         v16bf;
typedef __attribute__((ext_vector_type(2)))  __bf16         v2bf;
typedef __attribute__((ext_vector_type(16))) unsigned short v16u;
typedef __attribute__((ext_vector_type(8)))  unsigned short v8u;
typedef __attribute__((ext_vector_type(4)))  int            v4i;

#if __has_builtin(__builtin_amdgcn_global_load_async_to_lds_b128) && \
    __has_builtin(__builtin_amdgcn_s_wait_asynccnt)
#define HAVE_ASYNC_LDS 1
#else
#define HAVE_ASYNC_LDS 0
#endif

#if __has_builtin(__builtin_amdgcn_cvt_pk_bf16_f32)
#define HAVE_PK_BF16 1
#else
#define HAVE_PK_BF16 0
#endif

// f32 -> bf16 via native fptrunc (v_cvt_*bf16* when the target has it)
static __device__ __forceinline__ unsigned short cvt_bf16(float f) {
  __bf16 h = (__bf16)f;
  return __builtin_bit_cast(unsigned short, h);
}

// two f32 -> packed 2x bf16 in one dword
static __device__ __forceinline__ unsigned cvt2u(float a, float b) {
#if HAVE_PK_BF16
  v2bf p = __builtin_amdgcn_cvt_pk_bf16_f32(a, b);
  return __builtin_bit_cast(unsigned, p);
#else
  v2bf p; p[0] = (__bf16)a; p[1] = (__bf16)b;
  return __builtin_bit_cast(unsigned, p);
#endif
}

// 16B global -> LDS copy: async (gfx1250 ASYNCcnt path) or VGPR bounce.
static __device__ __forceinline__ void copy16_g2l(const unsigned short* src,
                                                  unsigned short* dst) {
#if HAVE_ASYNC_LDS
  __builtin_amdgcn_global_load_async_to_lds_b128((AS1 v4i*)src, (AS3 v4i*)dst, 0, 0);
#else
  *(v8u*)dst = *(const v8u*)src;
#endif
}

static __device__ __forceinline__ void wait_async_lds() {
#if HAVE_ASYNC_LDS
  __builtin_amdgcn_s_wait_asynccnt(0);
#endif
}

// Build a 16x32 bf16 WMMA operand fragment from row-major LDS (pitch LDP).
// Lanes 0-15 rows 0-15 hold K {0..7,16..23}; lanes 16-31 K {8..15,24..31}.
static __device__ __forceinline__ v16bf frag_from_lds(const unsigned short* rowbase, int koff) {
  v8u lo = *(const v8u*)(rowbase + koff);
  v8u hi = *(const v8u*)(rowbase + 16 + koff);
  v16u c;
#pragma unroll
  for (int i = 0; i < 8; ++i) { c[i] = lo[i]; c[i + 8] = hi[i]; }
  return __builtin_bit_cast(v16bf, c);
}

// ---------------------------------------------------------------- small init
__global__ void init_misc_kernel(int* counts, float* p_sum, float* lsesq) {
  int i = threadIdx.x;
  if (i < E_NUM) { counts[i] = 0; p_sum[i] = 0.f; }
  if (i == 0) lsesq[0] = 0.f;
}

__global__ void init_slots_kernel(int* slot_token, float* slot_gate, unsigned short* zbuf) {
  int i = blockIdx.x * 256 + threadIdx.x;
  if (i < R_PAD) { slot_token[i] = -1; slot_gate[i] = 0.f; }
  if (i < H_DIM) zbuf[i] = 0;
}

__global__ void y_init_kernel(const float* __restrict__ bias, float* __restrict__ y) {
  size_t i = (size_t)blockIdx.x * 256 + threadIdx.x;
  y[i] = bias[i & (H_DIM - 1)];
}

// x (fp32) -> xbf (bf16), packed conversion, 4 elements/thread
__global__ void xcvt_kernel(const float* __restrict__ x, unsigned short* __restrict__ xbf) {
  size_t i = ((size_t)blockIdx.x * 256 + threadIdx.x) * 4;
  float4 v = *(const float4*)(x + i);
  uint2 r;
  r.x = cvt2u(v.x, v.y);
  r.y = cvt2u(v.z, v.w);
  *(uint2*)(xbf + i) = r;
}

// ---------------------------------------------------------------- router
__global__ __launch_bounds__(256) void router_kernel(
    const float* __restrict__ x, const float* __restrict__ wr,
    int* __restrict__ top_idx, float* __restrict__ gates,
    int* __restrict__ counts, float* __restrict__ p_sum, float* __restrict__ lsesq) {
  __shared__ float blk_p[E_NUM];
  __shared__ float blk_z;
  const int tid = threadIdx.x, lane = tid & 31, wv = tid >> 5;
  const int t = blockIdx.x * 8 + wv;
  if (tid < E_NUM) blk_p[tid] = 0.f;
  if (tid == 0) blk_z = 0.f;
  __syncthreads();

  float acc[E_NUM];
#pragma unroll
  for (int e = 0; e < E_NUM; ++e) acc[e] = 0.f;
  const float* xr = x + (size_t)t * H_DIM;
  for (int i = 0; i < H_DIM / 32; ++i) {
    float xv = xr[i * 32 + lane];                     // 128B contiguous per wave
    const float* wrow = wr + (size_t)(i * 32 + lane) * E_NUM;
#pragma unroll
    for (int e = 0; e < E_NUM; ++e) acc[e] += xv * wrow[e];
  }
#pragma unroll
  for (int e = 0; e < E_NUM; ++e)
#pragma unroll
    for (int off = 16; off; off >>= 1) acc[e] += __shfl_xor(acc[e], off, 32);

  if (lane == 0) {
    float m = acc[0];
#pragma unroll
    for (int e = 1; e < E_NUM; ++e) m = fmaxf(m, acc[e]);
    float se = 0.f;
#pragma unroll
    for (int e = 0; e < E_NUM; ++e) se += expf(acc[e] - m);
    float lse = m + logf(se);
    float inv = 1.f / se;
#pragma unroll
    for (int e = 0; e < E_NUM; ++e) atomicAdd(&blk_p[e], expf(acc[e] - m) * inv);
    atomicAdd(&blk_z, lse * lse);
    int i0 = 0; float m0 = acc[0];
#pragma unroll
    for (int e = 1; e < E_NUM; ++e) if (acc[e] > m0) { m0 = acc[e]; i0 = e; }
    int i1 = -1; float m1 = -3.4e38f;
#pragma unroll
    for (int e = 0; e < E_NUM; ++e) if (e != i0 && acc[e] > m1) { m1 = acc[e]; i1 = e; }
    float g0 = 1.f / (1.f + expf(m1 - m0));
    top_idx[t * 2] = i0; top_idx[t * 2 + 1] = i1;
    gates[t * 2] = g0;   gates[t * 2 + 1] = 1.f - g0;
    atomicAdd(&counts[i0], 1); atomicAdd(&counts[i1], 1);
  }
  __syncthreads();
  if (tid < E_NUM) atomicAdd(&p_sum[tid], blk_p[tid]);
  if (tid == 0)    atomicAdd(lsesq, blk_z);
}

// ---------------------------------------------------------------- dispatch
__global__ void setup_tiles_kernel(const int* __restrict__ counts, int* cursor,
                                   int* tile_expert, int* tile_mbase, int* n_tiles) {
  if (threadIdx.x | blockIdx.x) return;
  int off = 0, tt = 0;
  for (int e = 0; e < E_NUM; ++e) {
    cursor[e] = off;
    int nt = (counts[e] + BM - 1) / BM;
    for (int j = 0; j < nt; ++j) { tile_expert[tt] = e; tile_mbase[tt] = off + j * BM; ++tt; }
    off += nt * BM;
  }
  *n_tiles = tt;
}

__global__ void scatter_kernel(const int* __restrict__ top_idx, const float* __restrict__ gates,
                               int* cursor, int* slot_token, float* slot_gate) {
  int i = blockIdx.x * 256 + threadIdx.x;
  if (i >= R_TOT) return;
  int e = top_idx[i];
  int s = atomicAdd(&cursor[e], 1);
  slot_token[s] = i >> 1;
  slot_gate[s]  = gates[i];
}

// ---------------------------------------------------------------- GEMM1 + GLU
// 128 rows x 64 cols of BOTH GLU halves of x @ w_in[e]; double-buffered LDS,
// async bf16 A staging, packed-convert B staging. G = silu(a)*g (bf16).
__global__ __launch_bounds__(256) void gemm1_glu_kernel(
    const unsigned short* __restrict__ xbf, const float* __restrict__ w_in,
    const int* __restrict__ tile_expert, const int* __restrict__ tile_mbase,
    const int* __restrict__ n_tiles, const int* __restrict__ slot_token,
    const unsigned short* __restrict__ zbuf, unsigned short* __restrict__ G) {
  if ((int)blockIdx.x >= *n_tiles) return;
  const int e      = tile_expert[blockIdx.x];
  const int m_base = tile_mbase[blockIdx.x];
  const int n0     = blockIdx.y * BN1;
  const float* __restrict__ W = w_in + (size_t)e * H_DIM * TWOF;

  __shared__ unsigned short a_lds[2][BM * LDP];
  __shared__ unsigned short b_lds[2][2][BN1 * LDP];

  const int tid = threadIdx.x, lane = tid & 31, wv = tid >> 5;
  const int wm = wv & 3, wn = wv >> 2;
  const int fr = lane & 15;
  const int koff = (lane >> 4) << 3;

  // per-thread staging pointers, hoisted out of the k loop
  const unsigned short* asrc[2]; int aoff[2];
#pragma unroll
  for (int p = 0; p < 2; ++p) {
    int lin = p * 256 + tid;              // 0..511 16B chunks (128 x 32 bf16)
    int row = lin >> 2, c8 = lin & 3;
    int tok = slot_token[m_base + row];
    asrc[p] = (tok >= 0) ? (xbf + (size_t)tok * H_DIM + c8 * 8) : (zbuf + c8 * 8);
    aoff[p] = row * LDP + c8 * 8;
  }
  const float* bsrc[2][2]; int boff[2][2];
#pragma unroll
  for (int h = 0; h < 2; ++h)
#pragma unroll
    for (int p = 0; p < 2; ++p) {
      int lin = p * 256 + tid;            // 0..511 float4s (32 x 64)
      int kr = lin >> 4, c4 = lin & 15;
      bsrc[h][p] = W + (h ? F_DIM : 0) + (size_t)kr * TWOF + n0 + c4 * 4;
      boff[h][p] = (c4 * 4) * LDP + kr;
    }

  v8f acc_a[2][2], acc_g[2][2];
#pragma unroll
  for (int i = 0; i < 2; ++i)
#pragma unroll
    for (int j = 0; j < 2; ++j) { acc_a[i][j] = (v8f){}; acc_g[i][j] = (v8f){}; }

  auto stage = [&](int kt, int buf) {
#pragma unroll
    for (int p = 0; p < 2; ++p)
      copy16_g2l(asrc[p] + kt * BK, &a_lds[buf][aoff[p]]);
#pragma unroll
    for (int h = 0; h < 2; ++h)
#pragma unroll
      for (int p = 0; p < 2; ++p) {
        float4 v = *(const float4*)(bsrc[h][p] + (size_t)kt * BK * TWOF);
        unsigned short* d = &b_lds[buf][h][boff[h][p]];
        unsigned lo = cvt2u(v.x, v.y), hi = cvt2u(v.z, v.w);
        d[0 * LDP] = (unsigned short)lo; d[1 * LDP] = (unsigned short)(lo >> 16);
        d[2 * LDP] = (unsigned short)hi; d[3 * LDP] = (unsigned short)(hi >> 16);
      }
  };

  stage(0, 0);
  const int NK = H_DIM / BK;
  for (int kt = 0; kt < NK; ++kt) {
    const int buf = kt & 1;
    wait_async_lds();
    __syncthreads();
    if (kt + 1 < NK) stage(kt + 1, buf ^ 1);

    v16bf afrag[2];
#pragma unroll
    for (int im = 0; im < 2; ++im)
      afrag[im] = frag_from_lds(&a_lds[buf][(wm * 32 + im * 16 + fr) * LDP], koff);
#pragma unroll
    for (int jn = 0; jn < 2; ++jn) {
      const int n = wn * 32 + jn * 16 + fr;
      v16bf bA = frag_from_lds(&b_lds[buf][0][n * LDP], koff);
      v16bf bG = frag_from_lds(&b_lds[buf][1][n * LDP], koff);
#pragma unroll
      for (int im = 0; im < 2; ++im) {
        acc_a[im][jn] = __builtin_amdgcn_wmma_f32_16x16x32_bf16(
            false, afrag[im], false, bA, (short)0, acc_a[im][jn], false, false);
        acc_g[im][jn] = __builtin_amdgcn_wmma_f32_16x16x32_bf16(
            false, afrag[im], false, bG, (short)0, acc_g[im][jn], false, false);
      }
    }
  }
  // GLU epilogue: C layout -> lane holds (r + 8*(lane>=16), lane%16)
  const int rlo = (lane >> 4) << 3;
#pragma unroll
  for (int im = 0; im < 2; ++im)
#pragma unroll
    for (int jn = 0; jn < 2; ++jn)
#pragma unroll
      for (int r = 0; r < 8; ++r) {
        int row = wm * 32 + im * 16 + rlo + r;
        int col = n0 + wn * 32 + jn * 16 + fr;
        float a = acc_a[im][jn][r], g = acc_g[im][jn][r];
        float s = (a / (1.f + __expf(-a))) * g;
        G[(size_t)(m_base + row) * F_DIM + col] = cvt_bf16(s);
      }
}

// ---------------------------------------------------------------- GEMM2 + combine
__global__ __launch_bounds__(256) void gemm2_scatter_kernel(
    const unsigned short* __restrict__ G, const float* __restrict__ w_out,
    const int* __restrict__ tile_expert, const int* __restrict__ tile_mbase,
    const int* __restrict__ n_tiles, const int* __restrict__ slot_token,
    const float* __restrict__ slot_gate, float* __restrict__ y) {
  if ((int)blockIdx.x >= *n_tiles) return;
  const int e      = tile_expert[blockIdx.x];
  const int m_base = tile_mbase[blockIdx.x];
  const int n0     = blockIdx.y * BN2;
  const float* __restrict__ Wo = w_out + (size_t)e * F_DIM * H_DIM;

  __shared__ unsigned short a_lds[2][BM * LDP];
  __shared__ unsigned short b_lds[2][BN2 * LDP];

  const int tid = threadIdx.x, lane = tid & 31, wv = tid >> 5;
  const int wm = wv & 3, wn = wv >> 2;
  const int fr = lane & 15;
  const int koff = (lane >> 4) << 3;

  const unsigned short* asrc[2]; int aoff[2];
#pragma unroll
  for (int p = 0; p < 2; ++p) {
    int lin = p * 256 + tid;              // 0..511 16B chunks (128 x 32 bf16)
    int row = lin >> 2, c8 = lin & 3;
    asrc[p] = G + (size_t)(m_base + row) * F_DIM + c8 * 8;
    aoff[p] = row * LDP + c8 * 8;
  }
  const float* bsrc[4]; int boff[4];
#pragma unroll
  for (int p = 0; p < 4; ++p) {
    int lin = p * 256 + tid;              // 0..1023 float4s (32 x 128)
    int kr = lin >> 5, c4 = lin & 31;
    bsrc[p] = Wo + (size_t)kr * H_DIM + n0 + c4 * 4;
    boff[p] = (c4 * 4) * LDP + kr;
  }

  v8f acc[2][4];
#pragma unroll
  for (int i = 0; i < 2; ++i)
#pragma unroll
    for (int j = 0; j < 4; ++j) acc[i][j] = (v8f){};

  auto stage = [&](int kt, int buf) {
#pragma unroll
    for (int p = 0; p < 2; ++p)
      copy16_g2l(asrc[p] + kt * BK, &a_lds[buf][aoff[p]]);
#pragma unroll
    for (int p = 0; p < 4; ++p) {
      float4 v = *(const float4*)(bsrc[p] + (size_t)kt * BK * H_DIM);
      unsigned short* d = &b_lds[buf][boff[p]];
      unsigned lo = cvt2u(v.x, v.y), hi = cvt2u(v.z, v.w);
      d[0 * LDP] = (unsigned short)lo; d[1 * LDP] = (unsigned short)(lo >> 16);
      d[2 * LDP] = (unsigned short)hi; d[3 * LDP] = (unsigned short)(hi >> 16);
    }
  };

  stage(0, 0);
  const int NK = F_DIM / BK;
  for (int kt = 0; kt < NK; ++kt) {
    const int buf = kt & 1;
    wait_async_lds();
    __syncthreads();
    if (kt + 1 < NK) stage(kt + 1, buf ^ 1);

    v16bf afrag[2];
#pragma unroll
    for (int im = 0; im < 2; ++im)
      afrag[im] = frag_from_lds(&a_lds[buf][(wm * 32 + im * 16 + fr) * LDP], koff);
#pragma unroll
    for (int jn = 0; jn < 4; ++jn) {
      v16bf bf = frag_from_lds(&b_lds[buf][(wn * 64 + jn * 16 + fr) * LDP], koff);
#pragma unroll
      for (int im = 0; im < 2; ++im)
        acc[im][jn] = __builtin_amdgcn_wmma_f32_16x16x32_bf16(
            false, afrag[im], false, bf, (short)0, acc[im][jn], false, false);
    }
  }
  const int rlo = (lane >> 4) << 3;
#pragma unroll
  for (int im = 0; im < 2; ++im)
#pragma unroll
    for (int jn = 0; jn < 4; ++jn)
#pragma unroll
      for (int r = 0; r < 8; ++r) {
        int row = wm * 32 + im * 16 + rlo + r;
        int slot = m_base + row;
        int tok = slot_token[slot];
        if (tok >= 0) {
          int col = n0 + wn * 64 + jn * 16 + fr;
          unsafeAtomicAdd(&y[(size_t)tok * H_DIM + col], acc[im][jn][r] * slot_gate[slot]);
        }
      }
}

// ---------------------------------------------------------------- loss
__global__ void loss_kernel(const int* __restrict__ counts, const float* __restrict__ p_sum,
                            const float* __restrict__ lsesq, float* __restrict__ y) {
  if (threadIdx.x | blockIdx.x) return;
  float ps = 0.f, cs = 0.f;
  for (int e = 0; e < E_NUM; ++e) { ps += p_sum[e]; cs += (float)counts[e]; }
  float dot = 0.f;
  for (int e = 0; e < E_NUM; ++e) dot += (p_sum[e] / ps) * ((float)counts[e] / cs);
  y[(size_t)T_TOK * H_DIM] = (float)E_NUM * dot + 0.1f * lsesq[0] / (float)T_TOK;
}

// ---------------------------------------------------------------- host
extern "C" void kernel_launch(void* const* d_in, const int* in_sizes, int n_in,
                              void* d_out, int out_size, void* d_ws, size_t ws_size,
                              hipStream_t stream) {
  const float* x     = (const float*)d_in[0];
  const float* wr    = (const float*)d_in[1];
  const float* w_in  = (const float*)d_in[2];
  const float* w_out = (const float*)d_in[3];
  const float* bias  = (const float*)d_in[4];
  float* y = (float*)d_out;

  char* ws = (char*)d_ws;                                      // ~44 MB total
  int*   top_idx     = (int*)  (ws);                           // 8192 i32
  float* gates       = (float*)(ws + 32768);                   // 8192 f32
  int*   counts      = (int*)  (ws + 65536);
  int*   cursor      = (int*)  (ws + 65536 + 64);
  float* p_sum       = (float*)(ws + 65536 + 128);
  float* lsesq       = (float*)(ws + 65536 + 192);
  int*   n_tiles     = (int*)  (ws + 65536 + 256);
  int*   tile_expert = (int*)  (ws + 65536 + 512);
  int*   tile_mbase  = (int*)  (ws + 65536 + 1024);
  int*   slot_token  = (int*)  (ws + 67584);                   // R_PAD i32
  float* slot_gate   = (float*)(ws + 67584 + R_PAD * 4);       // R_PAD f32
  unsigned short* zbuf = (unsigned short*)(ws + 141312);       // H_DIM bf16 zeros
  unsigned short* xbf  = (unsigned short*)(ws + 143360);       // T*H bf16
  unsigned short* G    = (unsigned short*)(ws + 143360 + (size_t)T_TOK * H_DIM * 2);

  init_misc_kernel<<<1, 32, 0, stream>>>(counts, p_sum, lsesq);
  router_kernel<<<T_TOK / 8, 256, 0, stream>>>(x, wr, top_idx, gates, counts, p_sum, lsesq);
  setup_tiles_kernel<<<1, 1, 0, stream>>>(counts, cursor, tile_expert, tile_mbase, n_tiles);
  init_slots_kernel<<<R_PAD / 256, 256, 0, stream>>>(slot_token, slot_gate, zbuf);
  scatter_kernel<<<R_TOT / 256, 256, 0, stream>>>(top_idx, gates, cursor, slot_token, slot_gate);
  xcvt_kernel<<<(T_TOK * H_DIM) / 1024, 256, 0, stream>>>(x, xbf);
  y_init_kernel<<<(T_TOK * H_DIM) / 256, 256, 0, stream>>>(bias, y);
  gemm1_glu_kernel<<<dim3(MAX_TILES, F_DIM / BN1), 256, 0, stream>>>(
      xbf, w_in, tile_expert, tile_mbase, n_tiles, slot_token, zbuf, G);
  gemm2_scatter_kernel<<<dim3(MAX_TILES, H_DIM / BN2), 256, 0, stream>>>(
      G, w_out, tile_expert, tile_mbase, n_tiles, slot_token, slot_gate, y);
  loss_kernel<<<1, 1, 0, stream>>>(counts, p_sum, lsesq, y);
}
